// BertForMultiLabelSequenceClassification_77945066488466
// MI455X (gfx1250) — matrix operand order repaired and checked by reading, not compiled, not run on hardware
//
#include <hip/hip_runtime.h>
#include <stdint.h>

#define N_NODES 4096
#define F_DIM   768
#define H_HEADS 6
#define D_DIM   128
#define NW      (N_NODES / 32)   // 128 words per bitmask column
#define ALPHA   0.2f

typedef __bf16 v16bf __attribute__((ext_vector_type(16)));
typedef __bf16 v8bf  __attribute__((ext_vector_type(8)));
typedef float  v8f   __attribute__((ext_vector_type(8)));

__device__ __forceinline__ unsigned short f2bf(float f) {
  union { float f; uint32_t u; } v; v.f = f;
  uint32_t u = v.u;
  uint32_t r = (u + 0x7FFFu + ((u >> 16) & 1u)) >> 16;  // RNE
  return (unsigned short)r;
}
__device__ __forceinline__ float bf2f(unsigned short h) {
  union { uint32_t u; float f; } v; v.u = ((uint32_t)h) << 16;
  return v.f;
}

// CDNA5 async global->LDS copy (ASYNCcnt-tracked), 16 bytes per lane.
// Generic LDS pointer's low 32 bits are the wave-relative LDS address.
__device__ __forceinline__ void async_b128(const unsigned short* lds_dst,
                                           const unsigned short* gsrc) {
  uint32_t l = (uint32_t)(uintptr_t)lds_dst;
  uint64_t g = (uint64_t)(uintptr_t)gsrc;
  asm volatile("global_load_async_to_lds_b128 %0, %1, off"
               :: "v"(l), "v"(g) : "memory");
}
__device__ __forceinline__ void wait_async_le4() {
  asm volatile("s_wait_asynccnt 0x4" ::: "memory");
}
__device__ __forceinline__ void wait_async_0() {
  asm volatile("s_wait_asynccnt 0x0" ::: "memory");
}

// ---------------------------------------------------------------- converts
__global__ void cvt_f32_bf16(const float* __restrict__ in,
                             unsigned short* __restrict__ out, int n) {
  int i = blockIdx.x * blockDim.x + threadIdx.x;
  if (i < n) out[i] = f2bf(in[i]);
}

// Column-major adjacency bitmask: bitT[c*NW + w] bit b = adj[32w+b][c]
__global__ void pack_adj_colmajor(const int* __restrict__ adj,
                                  uint32_t* __restrict__ bitT) {
  int tid = blockIdx.x * blockDim.x + threadIdx.x;   // tid = w*N + c
  int c = tid & (N_NODES - 1);
  int w = tid >> 12;
  if (w >= NW) return;
  uint32_t m = 0;
  for (int b = 0; b < 32; ++b)
    m |= (adj[(size_t)(32 * w + b) * N_NODES + c] > 0 ? 1u : 0u) << b;
  bitT[(size_t)c * NW + w] = m;
}

// ---------------------------------------------------------------- GEMM: Wh = A @ W  (bf16 in, bf16 out, f32 acc)
// A: [N][768] bf16, W: [H][768][128] bf16, Wh: [H][N][128] bf16
// Double-buffered LDS staged via GLOBAL_LOAD_ASYNC_TO_LDS_B128.
__global__ __launch_bounds__(256) void gemm_bf16_wmma(
    const unsigned short* __restrict__ A,
    const unsigned short* __restrict__ W,
    unsigned short* __restrict__ Wh) {
  __shared__ __align__(32) unsigned short lA[2][128 * 32];
  __shared__ __align__(32) unsigned short lB[2][32 * 128];
  const int h    = blockIdx.y;
  const int row0 = blockIdx.x * 128;
  const int tid  = threadIdx.x;
  const int wave = tid >> 5;
  const int lane = tid & 31;
  const int mrow = lane & 15;
  const int g    = (lane >> 4) * 8;

  v8f acc[8] = {};

  // issue this thread's 4 async b128 copies for K-tile k0 into buffer buf
  auto issue_tile = [&](int k0, int buf) {
#pragma unroll
    for (int it = 0; it < 2; ++it) {
      int idx = tid * 16 + it * 8;
      int ra = idx >> 5, ca = idx & 31;
      async_b128(&lA[buf][idx], &A[(size_t)(row0 + ra) * F_DIM + k0 + ca]);
      int rb = idx >> 7, cb = idx & 127;
      async_b128(&lB[buf][idx], &W[((size_t)h * F_DIM + k0 + rb) * D_DIM + cb]);
    }
  };

  issue_tile(0, 0);
  int kb = 0;
  for (int k0 = 0; k0 < F_DIM; k0 += 32, kb ^= 1) {
    if (k0 + 32 < F_DIM) {
      issue_tile(k0 + 32, kb ^ 1);   // prefetch next tile into other buffer
      wait_async_le4();              // in-order: waits only for current tile
    } else {
      wait_async_0();
    }
    __syncthreads();                 // current tile visible to all waves

    const unsigned short* ap = &lA[kb][(wave * 16 + mrow) * 32];
    v8bf alo = *(const v8bf*)&ap[g];
    v8bf ahi = *(const v8bf*)&ap[16 + g];
    v16bf afrag = __builtin_shufflevector(alo, ahi,
        0,1,2,3,4,5,6,7,8,9,10,11,12,13,14,15);
#pragma unroll
    for (int t = 0; t < 8; ++t) {
      v16bf bfrag;
#pragma unroll
      for (int s = 0; s < 16; ++s) {
        int k = (s < 8) ? (g + s) : (16 + g + s - 8);
        bfrag[s] = __builtin_bit_cast(__bf16, lB[kb][k * 128 + t * 16 + mrow]);
      }
      acc[t] = __builtin_amdgcn_wmma_f32_16x16x32_bf16(
          false, afrag, false, bfrag, (short)0, acc[t], false, false);
    }
    __syncthreads();                 // all reads done before buffer reuse
  }

  const int rbase = row0 + wave * 16 + ((lane >> 4) << 3);
#pragma unroll
  for (int t = 0; t < 8; ++t)
#pragma unroll
    for (int r = 0; r < 8; ++r)
      Wh[((size_t)h * N_NODES + rbase + r) * D_DIM + t * 16 + mrow] =
          f2bf(acc[t][r]);
}

// ---------------------------------------------------------------- s1/s2 = Wh . a
__global__ void s1s2_kernel(const unsigned short* __restrict__ Wh,
                            const float* __restrict__ a,      // [H][2D]
                            float* __restrict__ s1, float* __restrict__ s2) {
  int tid = blockIdx.x * blockDim.x + threadIdx.x;            // (h,n)
  int h = tid / N_NODES, n = tid % N_NODES;
  if (h >= H_HEADS) return;
  const unsigned short* wp = &Wh[((size_t)h * N_NODES + n) * D_DIM];
  const float* ap = &a[(size_t)h * 2 * D_DIM];
  float x1 = 0.f, x2 = 0.f;
#pragma unroll 8
  for (int d = 0; d < D_DIM; ++d) {
    float w = bf2f(wp[d]);
    x1 += w * ap[d];
    x2 += w * ap[D_DIM + d];
  }
  s1[tid] = x1; s2[tid] = x2;
}

// ---------------------------------------------------------------- exact row max: M_i = lrelu(s1_i + max_{j in adj(i)} s2_j)
__global__ __launch_bounds__(256) void rowmax_kernel(
    const int* __restrict__ adj, const float* __restrict__ s1,
    const float* __restrict__ s2, float* __restrict__ M) {
  int gi = blockIdx.x * 8 + (threadIdx.x >> 5);               // (h,i), wave each
  int lane = threadIdx.x & 31;
  int h = gi / N_NODES, i = gi % N_NODES;
  if (h >= H_HEADS) return;
  const float* s2p = &s2[(size_t)h * N_NODES];
  float m = -3.0e38f;
  for (int j = lane; j < N_NODES; j += 32)
    if (adj[(size_t)i * N_NODES + j] > 0) m = fmaxf(m, s2p[j]);
#pragma unroll
  for (int off = 16; off; off >>= 1) m = fmaxf(m, __shfl_xor(m, off, 32));
  if (lane == 0) {
    float e = s1[(size_t)h * N_NODES + i] + m;
    e = e > 0.f ? e : ALPHA * e;
    if (m < -1.0e38f) e = 0.f;                                // isolated row guard
    M[(size_t)h * N_NODES + i] = e;
  }
}

// ---------------------------------------------------------------- fused attention + aggregate + double ELU
// One wave per 16-row block per head; flash-style over 32-col tiles.
__global__ __launch_bounds__(128) void attn_kernel(
    const uint32_t* __restrict__ bitT,
    const float* __restrict__ s1, const float* __restrict__ s2,
    const float* __restrict__ M,
    const unsigned short* __restrict__ Wh,      // [H][N][128] bf16
    unsigned short* __restrict__ outBF,         // [N][768] bf16 (layer 1)
    float* __restrict__ outF) {                 // [N][768] f32  (layer 2)
  __shared__ __align__(32) unsigned short lP[4][16 * 32];
  const int h    = blockIdx.y;
  const int wave = threadIdx.x >> 5;
  const int lane = threadIdx.x & 31;
  const int row0 = (blockIdx.x * 4 + wave) * 16;
  const int mrow = lane & 15;
  const int hi   = lane >> 4;
  const int g    = hi * 8;

  // broadcast per-row constants across the wave
  float s1mine = s1[(size_t)h * N_NODES + row0 + mrow];
  float Mmine  = M [(size_t)h * N_NODES + row0 + mrow];
  float s1r[16], Mr[16];
#pragma unroll
  for (int r = 0; r < 16; ++r) {
    s1r[r] = __shfl(s1mine, r, 32);
    Mr[r]  = __shfl(Mmine,  r, 32);
  }

  v8f acc[8] = {};
  float lsum[16];
#pragma unroll
  for (int r = 0; r < 16; ++r) lsum[r] = 0.f;

  unsigned short* myP = lP[wave];
  const int wword  = row0 >> 5;
  const int wshift = row0 & 31;

  for (int j0 = 0; j0 < N_NODES; j0 += 32) {
    const int c = j0 + lane;
    const float s2c = s2[(size_t)h * N_NODES + c];
    const uint32_t aw = bitT[(size_t)c * NW + wword] >> wshift;
    if (j0 + 32 < N_NODES)    // prefetch next Wh tile (global_prefetch_b8)
      __builtin_prefetch(&Wh[((size_t)h * N_NODES + j0 + 32 + lane) * D_DIM], 0, 0);
#pragma unroll
    for (int r = 0; r < 16; ++r) {
      float p = 0.f;
      if ((aw >> r) & 1u) {
        float e = s1r[r] + s2c;
        e = e > 0.f ? e : ALPHA * e;
        p = __expf(e - Mr[r]);                  // exact: Mr is true row max
      }
      lsum[r] += p;
      myP[r * 32 + lane] = f2bf(p);             // lane = column
    }
    // intra-wave LDS RAW: DS ops are in-order per wave; compiler inserts waits
    const unsigned short* ap = &myP[mrow * 32];
    v8bf alo = *(const v8bf*)&ap[g];
    v8bf ahi = *(const v8bf*)&ap[16 + g];
    v16bf afrag = __builtin_shufflevector(alo, ahi,
        0,1,2,3,4,5,6,7,8,9,10,11,12,13,14,15);
    const unsigned short* bp = &Wh[((size_t)h * N_NODES + j0) * D_DIM + mrow];
#pragma unroll
    for (int t = 0; t < 8; ++t) {
      v16bf bfrag;
#pragma unroll
      for (int s = 0; s < 16; ++s) {
        int k = (s < 8) ? (g + s) : (16 + g + s - 8);
        bfrag[s] = __builtin_bit_cast(__bf16, bp[k * D_DIM + t * 16]);
      }
      acc[t] = __builtin_amdgcn_wmma_f32_16x16x32_bf16(
          false, afrag, false, bfrag, (short)0, acc[t], false, false);
    }
  }

  // reduce softmax denominators across lanes
  float ltot[16];
#pragma unroll
  for (int r = 0; r < 16; ++r) {
    float v = lsum[r];
#pragma unroll
    for (int off = 16; off; off >>= 1) v += __shfl_xor(v, off, 32);
    ltot[r] = v;
  }

#pragma unroll
  for (int t = 0; t < 8; ++t)
#pragma unroll
    for (int r = 0; r < 8; ++r) {
      float l  = hi ? ltot[r + 8] : ltot[r];
      float hp = acc[t][r] / fmaxf(l, 1e-30f);
      // elu(elu(x)): x>=0 -> x ; x<0 -> exp(exp(x)-1)-1
      float o = hp >= 0.f ? hp : (__expf(__expf(hp) - 1.f) - 1.f);
      int row = row0 + r + 8 * hi;
      int col = t * 16 + mrow;
      size_t oidx = (size_t)row * (H_HEADS * D_DIM) + (size_t)h * D_DIM + col;
      if (outF) outF[oidx] = o;
      else      outBF[oidx] = f2bf(o);
    }
}

// ---------------------------------------------------------------- host
extern "C" void kernel_launch(void* const* d_in, const int* in_sizes, int n_in,
                              void* d_out, int out_size, void* d_ws, size_t ws_size,
                              hipStream_t stream) {
  (void)in_sizes; (void)n_in; (void)out_size; (void)ws_size;
  const float* x   = (const float*)d_in[0];
  const int*   adj = (const int*)d_in[1];
  const float* W1  = (const float*)d_in[2];
  const float* a1  = (const float*)d_in[3];
  const float* W2  = (const float*)d_in[4];
  const float* a2  = (const float*)d_in[5];
  float* out = (float*)d_out;

  char* ws = (char*)d_ws;
  size_t off = 0;
  auto alloc = [&](size_t bytes) -> void* {
    void* p = ws + off; off += (bytes + 255) & ~(size_t)255; return p;
  };
  uint32_t*       bitT = (uint32_t*)      alloc((size_t)N_NODES * NW * 4);        // 2 MB
  unsigned short* xbf  = (unsigned short*)alloc((size_t)N_NODES * F_DIM * 2);     // 6 MB
  unsigned short* w1bf = (unsigned short*)alloc((size_t)H_HEADS * F_DIM * D_DIM * 2);
  unsigned short* w2bf = (unsigned short*)alloc((size_t)H_HEADS * F_DIM * D_DIM * 2);
  unsigned short* Wh   = (unsigned short*)alloc((size_t)H_HEADS * N_NODES * D_DIM * 2);
  float*          s1   = (float*)         alloc((size_t)H_HEADS * N_NODES * 4);
  float*          s2   = (float*)         alloc((size_t)H_HEADS * N_NODES * 4);
  float*          Mx   = (float*)         alloc((size_t)H_HEADS * N_NODES * 4);
  unsigned short* h1bf = (unsigned short*)alloc((size_t)N_NODES * F_DIM * 2);     // 6 MB

  const int HN = H_HEADS * N_NODES;

  pack_adj_colmajor<<<(N_NODES * NW) / 256, 256, 0, stream>>>(adj, bitT);
  cvt_f32_bf16<<<(N_NODES * F_DIM + 255) / 256, 256, 0, stream>>>(x, xbf, N_NODES * F_DIM);
  cvt_f32_bf16<<<(H_HEADS * F_DIM * D_DIM + 255) / 256, 256, 0, stream>>>(W1, w1bf, H_HEADS * F_DIM * D_DIM);
  cvt_f32_bf16<<<(H_HEADS * F_DIM * D_DIM + 255) / 256, 256, 0, stream>>>(W2, w2bf, H_HEADS * F_DIM * D_DIM);

  // ---- layer 1
  gemm_bf16_wmma<<<dim3(N_NODES / 128, H_HEADS), 256, 0, stream>>>(xbf, w1bf, Wh);
  s1s2_kernel<<<HN / 256, 256, 0, stream>>>(Wh, a1, s1, s2);
  rowmax_kernel<<<HN / 8, 256, 0, stream>>>(adj, s1, s2, Mx);
  attn_kernel<<<dim3(N_NODES / 64, H_HEADS), 128, 0, stream>>>(
      bitT, s1, s2, Mx, Wh, h1bf, nullptr);

  // ---- layer 2
  gemm_bf16_wmma<<<dim3(N_NODES / 128, H_HEADS), 256, 0, stream>>>(h1bf, w2bf, Wh);
  s1s2_kernel<<<HN / 256, 256, 0, stream>>>(Wh, a2, s1, s2);
  rowmax_kernel<<<HN / 8, 256, 0, stream>>>(adj, s1, s2, Mx);
  attn_kernel<<<dim3(N_NODES / 64, H_HEADS), 128, 0, stream>>>(
      bitT, s1, s2, Mx, Wh, nullptr, out);
}
